// GkNN_86285892977149
// MI455X (gfx1250) — matrix operand order repaired and checked by reading, not compiled, not run on hardware
//
#include <hip/hip_runtime.h>
#include <hip/hip_bf16.h>

// ---------------------------------------------------------------------------
// GkNN (Galerkin transformer neural operator) on MI455X / gfx1250.
// All GEMMs run on v_wmma_f32_16x16x32_bf16 (bf16 in, f32 accumulate).
// Activations stay in (b, n, c) row-major f32; LDS stages bf16 tiles.
// ---------------------------------------------------------------------------

typedef __attribute__((ext_vector_type(16))) __bf16        v16bf;
typedef __attribute__((ext_vector_type(8)))  float         v8f;
typedef __attribute__((ext_vector_type(8)))  unsigned short us8;

__device__ __forceinline__ unsigned short f2bf(float f) {
    unsigned u = __builtin_bit_cast(unsigned, f);
    u += 0x7FFFu + ((u >> 16) & 1u);          // round-to-nearest-even
    return (unsigned short)(u >> 16);
}
__device__ __forceinline__ float bf2f(unsigned short h) {
    return __builtin_bit_cast(float, ((unsigned)h) << 16);
}
__device__ __forceinline__ float gelu(float v) {
    return 0.5f * v * (1.0f + erff(v * 0.70710678118654752f));
}

// A-fragment (16x32 bf16): lane<16 holds row M=lane, K = 0..15 contiguous;
// lane>=16 holds row M=lane-16, K = 16..31 contiguous.  p points at K-start.
__device__ __forceinline__ v16bf ldfragA(const unsigned short* p) {
    struct S { us8 a, b; } s;
    s.a = *reinterpret_cast<const us8*>(p);
    s.b = *reinterpret_cast<const us8*>(p + 8);
    return __builtin_bit_cast(v16bf, s);
}
// B-fragment (32x16 bf16): lane<16 holds col N=lane, K = 0..15 contiguous;
// lane>=16 holds col N=lane-16, K = 16..31 contiguous.  p points at K-start.
__device__ __forceinline__ v16bf ldfragB(const unsigned short* p) {
    struct S { us8 a, b; } s;
    s.a = *reinterpret_cast<const us8*>(p);
    s.b = *reinterpret_cast<const us8*>(p + 8);
    return __builtin_bit_cast(v16bf, s);
}
__device__ __forceinline__ v8f wmma_bf16(v16bf a, v16bf b, v8f c) {
    return __builtin_amdgcn_wmma_f32_16x16x32_bf16(
        false, a, false, b, (short)0, c, false, false);
}
__device__ __forceinline__ v8f zero8() {
    v8f v;
    #pragma unroll
    for (int i = 0; i < 8; ++i) v[i] = 0.f;
    return v;
}

// ---------------------------------------------------------------------------
// Lift: t[r][d] = gelu(x[r][0]*w1[0][d] + x[r][1]*w1[1][d] + b1[d])
// ---------------------------------------------------------------------------
__global__ void k_lift(const float* __restrict__ x, const float* __restrict__ w1,
                       const float* __restrict__ b1, float* __restrict__ out,
                       long long total) {
    long long idx = (long long)blockIdx.x * blockDim.x + threadIdx.x;
    if (idx >= total) return;
    long long r = idx >> 7; int d = (int)(idx & 127);
    float v = x[r * 2] * w1[d] + x[r * 2 + 1] * w1[128 + d] + b1[d];
    out[idx] = gelu(v);
}

// ---------------------------------------------------------------------------
// NN GEMM:  C[b][r][n] = post + act( A[b][r][:] @ W[b][:][n] + bias + pre )
// A: (rows,128) f32 per batch, W: 128x128 f32 (w_trans: stored (n,k)).
// Block = 64(M) x 128(N), 8 waves of 32x32 (2x2 wmma frags), Kc = 128.
// ---------------------------------------------------------------------------
#define BM 64
#define WST 136   // LDS row stride (halves): 272B rows, 16B aligned

__global__ __launch_bounds__(256) void k_gemm_nn(
    const float* __restrict__ A, long long a_bs,
    const float* __restrict__ W, long long w_bs, int w_trans,
    const float* __restrict__ bias,
    const float* __restrict__ pre,  long long pre_bs,
    const float* __restrict__ post, long long post_bs,
    float* __restrict__ C, long long c_bs, int act)
{
    __shared__ unsigned short As[BM][WST];
    __shared__ unsigned short Wt[128][WST];
    const int tid = threadIdx.x;
    const int b = blockIdx.z;
    const long long row0 = (long long)blockIdx.x * BM;
    const float* Ab = A + a_bs * b;
    const float* Wb = W + w_bs * b;

    if (w_trans) {  // W stored (n,k): linear read, contiguous LDS write
        for (int i = tid; i < 128 * 128; i += 256) {
            int n = i >> 7, k = i & 127; Wt[n][k] = f2bf(Wb[i]);
        }
    } else {        // W stored (k,n): linear read, strided LDS write
        for (int i = tid; i < 128 * 128; i += 256) {
            int k = i >> 7, n = i & 127; Wt[n][k] = f2bf(Wb[i]);
        }
    }
    for (int i = tid; i < BM * 128; i += 256) {
        int r = i >> 7, c = i & 127;
        As[r][c] = f2bf(Ab[(row0 + r) * 128 + c]);
    }
    __syncthreads();

    const int lane = tid & 31;
    const int wave = tid >> 5;
    const int wm = (wave >> 2) * 32;   // 0 / 32
    const int wn = (wave & 3) * 32;    // 0,32,64,96
    const int fr = lane & 15;
    const int ah = (lane >> 4) << 4;   // A half-offset: 0 or 16
    const int bh = (lane >> 4) << 4;   // B half-offset: 0 or 16

    v8f acc[2][2];
    #pragma unroll
    for (int i = 0; i < 2; ++i)
        #pragma unroll
        for (int j = 0; j < 2; ++j) acc[i][j] = zero8();

    #pragma unroll
    for (int kk = 0; kk < 128; kk += 32) {
        v16bf af[2], bf[2];
        #pragma unroll
        for (int mi = 0; mi < 2; ++mi)
            af[mi] = ldfragA(&As[wm + mi * 16 + fr][kk + ah]);
        #pragma unroll
        for (int ni = 0; ni < 2; ++ni)
            bf[ni] = ldfragB(&Wt[wn + ni * 16 + fr][kk + bh]);
        #pragma unroll
        for (int mi = 0; mi < 2; ++mi)
            #pragma unroll
            for (int ni = 0; ni < 2; ++ni)
                acc[mi][ni] = wmma_bf16(af[mi], bf[ni], acc[mi][ni]);
    }

    const int mh = (lane >> 4) * 8;
    #pragma unroll
    for (int mi = 0; mi < 2; ++mi) {
        #pragma unroll
        for (int ni = 0; ni < 2; ++ni) {
            #pragma unroll
            for (int vr = 0; vr < 8; ++vr) {
                int m = wm + mi * 16 + vr + mh;
                int n = wn + ni * 16 + fr;
                long long r = row0 + m;
                float v = acc[mi][ni][vr];
                if (bias) v += bias[n];
                if (pre)  v += pre[pre_bs * b + r * 128 + n];
                if (act)  v = gelu(v);
                if (post) v += post[post_bs * b + r * 128 + n];
                C[c_bs * b + r * 128 + n] = v;
            }
        }
    }
}

// ---------------------------------------------------------------------------
// TN GEMM (long-K):  C[b][m][n] += sum_n' A[b][n'][m] * Bm[n'][n]
// Per block: one (batch, K-split) pair, full 128x128 output, split-K via
// global_atomic_add_f32.  Operands transposed f32->bf16 into LDS per 32-slab.
// ---------------------------------------------------------------------------
#define TST 40    // LDS row stride (halves): 80B rows, 16B aligned

__global__ __launch_bounds__(256) void k_gemm_tn(
    const float* __restrict__ A, long long a_bs,
    const float* __restrict__ Bm,
    float* __restrict__ C, int n_chunk)
{
    __shared__ unsigned short At[128][TST];
    __shared__ unsigned short Bt[128][TST];
    const int tid = threadIdx.x;
    const int b = blockIdx.z;
    const float* Ab = A + a_bs * b;
    const int n0_base = blockIdx.x * n_chunk;
    const int n0_end  = n0_base + n_chunk;

    const int lane = tid & 31;
    const int wave = tid >> 5;
    const int wm = (wave >> 1) * 32;   // 0..96
    const int wn = (wave & 1) * 64;    // 0 / 64
    const int fr = lane & 15;
    const int ah = (lane >> 4) << 4;
    const int bh = (lane >> 4) << 4;

    v8f acc[2][4];
    #pragma unroll
    for (int i = 0; i < 2; ++i)
        #pragma unroll
        for (int j = 0; j < 4; ++j) acc[i][j] = zero8();

    for (int n0 = n0_base; n0 < n0_end; n0 += 32) {
        for (int i = tid; i < 32 * 128; i += 256) {
            int nn = i >> 7, c = i & 127;
            At[c][nn] = f2bf(Ab[(long long)(n0 + nn) * 128 + c]);
            Bt[c][nn] = f2bf(Bm[(long long)(n0 + nn) * 128 + c]);
        }
        if (n0 + 32 < n0_end) {   // prefetch next 16KB slab of each operand
            __builtin_prefetch(Ab + (long long)(n0 + 32) * 128 + tid * 16, 0, 1);
            __builtin_prefetch(Bm + (long long)(n0 + 32) * 128 + tid * 16, 0, 1);
        }
        __syncthreads();

        v16bf af[2], bf[4];
        #pragma unroll
        for (int mi = 0; mi < 2; ++mi)
            af[mi] = ldfragA(&At[wm + mi * 16 + fr][ah]);
        #pragma unroll
        for (int ni = 0; ni < 4; ++ni)
            bf[ni] = ldfragB(&Bt[wn + ni * 16 + fr][bh]);
        #pragma unroll
        for (int mi = 0; mi < 2; ++mi)
            #pragma unroll
            for (int ni = 0; ni < 4; ++ni)
                acc[mi][ni] = wmma_bf16(af[mi], bf[ni], acc[mi][ni]);
        __syncthreads();
    }

    const int mh = (lane >> 4) * 8;
    #pragma unroll
    for (int mi = 0; mi < 2; ++mi)
        #pragma unroll
        for (int ni = 0; ni < 4; ++ni)
            #pragma unroll
            for (int vr = 0; vr < 8; ++vr) {
                int m = wm + mi * 16 + vr + mh;
                int n = wn + ni * 16 + fr;
                atomicAdd(&C[((long long)b * 128 + m) * 128 + n],
                          acc[mi][ni][vr]);
            }
}

// ---------------------------------------------------------------------------
// Per-mode mixing:  xh2[b][o][k] = sum_i xh[b][i][k] * cw[i][o][k]
// One block per mode k; cw slice staged bf16 in LDS. (0.13 GFLOP total.)
// ---------------------------------------------------------------------------
__global__ __launch_bounds__(256) void k_mix(
    const float* __restrict__ xh, const float* __restrict__ cw,
    float* __restrict__ xh2)
{
    __shared__ float xs[2048];
    __shared__ unsigned short ws[128][130];
    const int k = blockIdx.x;
    const int tid = threadIdx.x;
    for (int i = tid; i < 2048; i += 256)  xs[i] = xh[(long long)i * 128 + k];
    for (int i = tid; i < 16384; i += 256) {
        int ii = i >> 7, o = i & 127;
        ws[ii][o] = f2bf(cw[(long long)i * 128 + k]);
    }
    __syncthreads();
    for (int j = tid; j < 2048; j += 256) {
        int bb = j >> 7, o = j & 127;
        const float* xb = &xs[bb * 128];
        float s = 0.f;
        #pragma unroll 4
        for (int ii = 0; ii < 128; ++ii) s += xb[ii] * bf2f(ws[ii][o]);
        xh2[(long long)j * 128 + k] = s;
    }
}

// ---------------------------------------------------------------------------
// Head:  out[r] = dot(t[r][:], w) + b   (one wave per row, shfl reduce)
// ---------------------------------------------------------------------------
__global__ __launch_bounds__(256) void k_fc2(
    const float* __restrict__ t, const float* __restrict__ w,
    const float* __restrict__ bsc, float* __restrict__ out, int rows)
{
    int row = blockIdx.x * 8 + (threadIdx.x >> 5);
    int lane = threadIdx.x & 31;
    if (row >= rows) return;
    const float4 tv = *reinterpret_cast<const float4*>(t + (long long)row * 128 + lane * 4);
    const float4 wv = *reinterpret_cast<const float4*>(w + lane * 4);
    float s = tv.x * wv.x + tv.y * wv.y + tv.z * wv.z + tv.w * wv.w;
    #pragma unroll
    for (int off = 16; off > 0; off >>= 1) s += __shfl_down(s, off, 32);
    if (lane == 0) out[row] = s + bsc[0];
}

// ---------------------------------------------------------------------------
extern "C" void kernel_launch(void* const* d_in, const int* in_sizes, int n_in,
                              void* d_out, int out_size, void* d_ws, size_t ws_size,
                              hipStream_t stream) {
    (void)in_sizes; (void)n_in; (void)out_size; (void)ws_size;
    const float* x      = (const float*)d_in[0];
    const float* Bmat   = (const float*)d_in[1];
    const float* fc0_w1 = (const float*)d_in[2];
    const float* fc0_b1 = (const float*)d_in[3];
    const float* fc0_w2 = (const float*)d_in[4];
    const float* fc0_b2 = (const float*)d_in[5];
    const float* a0_wk  = (const float*)d_in[6];
    const float* a0_bk  = (const float*)d_in[7];
    const float* a0_f1w = (const float*)d_in[8];
    const float* a0_f1b = (const float*)d_in[9];
    const float* a0_f2w = (const float*)d_in[10];
    const float* a0_f2b = (const float*)d_in[11];
    const float* c1_w   = (const float*)d_in[12];
    const float* w1_w   = (const float*)d_in[13];
    const float* w1_b   = (const float*)d_in[14];
    const float* a2_wk  = (const float*)d_in[15];
    const float* a2_bk  = (const float*)d_in[16];
    const float* a2_f1w = (const float*)d_in[17];
    const float* a2_f1b = (const float*)d_in[18];
    const float* a2_f2w = (const float*)d_in[19];
    const float* a2_f2b = (const float*)d_in[20];
    const float* c3_w   = (const float*)d_in[21];
    const float* w3_w   = (const float*)d_in[22];
    const float* w3_b   = (const float*)d_in[23];
    const float* fc1_w  = (const float*)d_in[24];
    const float* fc1_b  = (const float*)d_in[25];
    const float* fc2_w  = (const float*)d_in[26];
    const float* fc2_b  = (const float*)d_in[27];

    const long long R  = 16LL * 8192;   // 131072 rows
    const long long Rb = 8192;          // rows per batch
    char* wsb = (char*)d_ws;
    float* h   = (float*)wsb;                              // 64MB activation
    float* t1  = (float*)(wsb + (size_t)R * 128 * 4);      // 64MB activation
    float* s0  = (float*)(wsb + (size_t)2 * R * 128 * 4);  // 1MB  score / xh
    float* s1  = s0 + 16 * 128 * 128;                      // 1MB  xh2

    const dim3 blk(256);
    const size_t modeBytes = (size_t)16 * 128 * 128 * sizeof(float);

    // lift + fc0
    k_lift<<<dim3((unsigned)((R * 128 + 255) / 256)), blk, 0, stream>>>(
        x, fc0_w1, fc0_b1, t1, R * 128);
    k_gemm_nn<<<dim3((unsigned)(R / BM), 1, 1), blk, 0, stream>>>(
        t1, 0, fc0_w2, 0, 0, fc0_b2, nullptr, 0, nullptr, 0, h, 0, 0);

    // DoubleAttention: input h -> output t1
    auto attention = [&](const float* wk, const float* bk,
                         const float* f1w, const float* f1b,
                         const float* f2w, const float* f2b) {
        // key = h @ wk + bk
        k_gemm_nn<<<dim3((unsigned)(R / BM), 1, 1), blk, 0, stream>>>(
            h, 0, wk, 0, 0, bk, nullptr, 0, nullptr, 0, t1, 0, 0);
        // score_b = key_b^T @ Bmat   (split-K 16, atomic f32 reduce)
        (void)hipMemsetAsync(s0, 0, modeBytes, stream);
        k_gemm_tn<<<dim3(16, 1, 16), blk, 0, stream>>>(t1, Rb * 128, Bmat, s0, (int)(Rb / 16));
        // t1 = gelu(Bmat @ score_b) + h
        k_gemm_nn<<<dim3((unsigned)(Rb / BM), 1, 16), blk, 0, stream>>>(
            Bmat, 0, s0, 128 * 128, 0, nullptr, nullptr, 0, h, Rb * 128, t1, Rb * 128, 1);
        // h = gelu(t1 @ f1w + f1b);  t1 = t1 + (h @ f2w + f2b)
        k_gemm_nn<<<dim3((unsigned)(R / BM), 1, 1), blk, 0, stream>>>(
            t1, 0, f1w, 0, 0, f1b, nullptr, 0, nullptr, 0, h, 0, 1);
        k_gemm_nn<<<dim3((unsigned)(R / BM), 1, 1), blk, 0, stream>>>(
            h, 0, f2w, 0, 0, f2b, nullptr, 0, t1, 0, t1, 0, 0);
    };

    // GalerkinConv + Conv1d branch: input t1 -> output h
    auto galerkin = [&](const float* cw, const float* ww, const float* wb, int act) {
        (void)hipMemsetAsync(s0, 0, modeBytes, stream);
        k_gemm_tn<<<dim3(16, 1, 16), blk, 0, stream>>>(t1, Rb * 128, Bmat, s0, (int)(Rb / 16));
        k_mix<<<dim3(128), blk, 0, stream>>>(s0, cw, s1);
        // x2 = t1 @ ww + wb   -> h
        k_gemm_nn<<<dim3((unsigned)(R / BM), 1, 1), blk, 0, stream>>>(
            t1, 0, ww, 0, 0, wb, nullptr, 0, nullptr, 0, h, 0, 0);
        // h = t1 + act(Bmat @ xh2_b + x2)
        k_gemm_nn<<<dim3((unsigned)(Rb / BM), 1, 16), blk, 0, stream>>>(
            Bmat, 0, s1, 128 * 128, 1, nullptr, h, Rb * 128, t1, Rb * 128, h, Rb * 128, act);
    };

    attention(a0_wk, a0_bk, a0_f1w, a0_f1b, a0_f2w, a0_f2b);  // h  -> t1
    galerkin(c1_w, w1_w, w1_b, 1);                            // t1 -> h
    attention(a2_wk, a2_bk, a2_f1w, a2_f1b, a2_f2w, a2_f2b);  // h  -> t1
    galerkin(c3_w, w3_w, w3_b, 0);                            // t1 -> h

    // head
    k_gemm_nn<<<dim3((unsigned)(R / BM), 1, 1), blk, 0, stream>>>(
        h, 0, fc1_w, 0, 0, fc1_b, nullptr, 0, nullptr, 0, t1, 0, 1);
    k_fc2<<<dim3((unsigned)(R / 8)), blk, 0, stream>>>(
        t1, fc2_w, fc2_b, (float*)d_out, (int)R);
}
